// RHAN_79267916415236
// MI455X (gfx1250) — matrix operand order repaired and checked by reading, not compiled, not run on hardware
//
#include <hip/hip_runtime.h>
#include <hip/hip_bf16.h>

typedef __attribute__((ext_vector_type(16))) __bf16        v16bf;
typedef __attribute__((ext_vector_type(8)))  float         v8f;
typedef __attribute__((ext_vector_type(16))) unsigned short v16u;

constexpr int D  = 64;     // embedding dim
constexpr int C  = 9;      // categories (attention heads)
constexpr int H  = 16;     // MLP hidden (== WMMA tile dim)
constexpr int L  = 200;    // history length
constexpr int LP = 224;    // padded: 14 tiles of 16 rows, 7 K-steps of 32
constexpr int NT = LP / 16;   // 14
constexpr int KT = LP / 32;   // 7
constexpr int NTHREADS = C * 32;   // 9 waves, one per category

union Frag { uint4 q[2]; v16u v; };

__device__ __forceinline__ unsigned short f2bf(float f) {
  unsigned x = __float_as_uint(f);
  return (unsigned short)((x + 0x7fffu + ((x >> 16) & 1u)) >> 16);  // RNE
}
__device__ __forceinline__ float bf2f(unsigned short s) {
  return __uint_as_float(((unsigned)s) << 16);
}

__global__ __launch_bounds__(NTHREADS)
void din_fused_kernel(const int* __restrict__ user_in,
                      const int* __restrict__ rec_in,
                      const int* __restrict__ item_in,
                      const int* __restrict__ item_cat,
                      const float* __restrict__ user_emb,
                      const float* __restrict__ item_emb,
                      const float* __restrict__ W1,  const float* __restrict__ b1,
                      const float* __restrict__ W2,  const float* __restrict__ b2,
                      const float* __restrict__ Wt1, const float* __restrict__ bt1,
                      const float* __restrict__ Wt2, const float* __restrict__ bt2,
                      float* __restrict__ out)
{
  __shared__ __align__(16) unsigned short eb[LP * D];    // history emb [l][d], bf16
  __shared__ __align__(16) unsigned short ebT[D * LP];   // transposed  [d][l], bf16
  __shared__ __align__(16) unsigned short wb[16 * LP];   // softmax w   [cat][l], bf16 (rows 9..15 = 0)
  __shared__ int   catl[LP];                             // per-item category (-1 invalid)
  __shared__ float u_sh[D];                              // user embedding
  __shared__ float ubias[C * H];                         // b1 + W1[:,:,:D] @ u
  __shared__ __align__(16) float s_sh[C][LP];            // attention logits
  __shared__ float gvec[C][D];                           // group vectors
  __shared__ float s2_sh[16];
  __shared__ int   has_sh[16];
  __shared__ float w2_sh[16];

  const int b    = blockIdx.x;
  const int tid  = threadIdx.x;
  const int wid  = tid >> 5;       // 0..8 == category this wave owns
  const int lane = tid & 31;
  const int cat  = wid;
  const int hl   = lane & 15;      // 16-lane half index
  const int kb8  = (lane >= 16) ? 8 : 0;    // A-matrix K interleave
  const int kb16 = (lane >= 16) ? 16 : 0;   // unused (B fragments use kb8)

  // ---------------- phase 1: stage user row + history (f32 -> bf16 ×2) -----
  if (tid < D) u_sh[tid] = user_emb[(long)user_in[b] * D + tid];
  for (int l = wid; l < LP; l += C) {
    int r = (l < L) ? rec_in[b * L + l] : -1;
    float2 v = make_float2(0.f, 0.f);
    if (r >= 0) v = *(const float2*)(item_emb + (long)r * D + 2 * lane);
    const unsigned short bx = f2bf(v.x), by = f2bf(v.y);
    eb[l * D + 2 * lane]       = bx;
    eb[l * D + 2 * lane + 1]   = by;
    ebT[(2 * lane) * LP + l]     = bx;
    ebT[(2 * lane + 1) * LP + l] = by;
    if (lane == 0) catl[l] = (r >= 0) ? item_cat[r] : -1;
  }
  for (int i = tid; i < (16 - C) * LP; i += NTHREADS)    // zero pad rows 9..15 of wb
    wb[C * LP + i] = 0;

  // Loop-invariant A fragments for scores: A[m=n_hidden][k=d] = W1[cat, n, D+k]
  v16bf af0, af1;
  {
    const float* wrow = W1 + (long)(cat * H + hl) * (2 * D);
    v16u t0, t1;
#pragma unroll
    for (int j = 0; j < 8; ++j) {
      t0[j]     = f2bf(wrow[D + kb8 + j]);
      t0[j + 8] = f2bf(wrow[D + kb8 + 16 + j]);
      t1[j]     = f2bf(wrow[D + 32 + kb8 + j]);
      t1[j + 8] = f2bf(wrow[D + 32 + kb8 + 16 + j]);
    }
    af0 = __builtin_bit_cast(v16bf, t0);
    af1 = __builtin_bit_cast(v16bf, t1);
  }
  __syncthreads();

  // ---------------- phase 2: per-user column bias = b1 + W1[:,:,:D] @ u ----
  if (tid < C * H) {
    const float* wrow = W1 + (long)tid * (2 * D);
    float acc = b1[tid];
#pragma unroll 8
    for (int i = 0; i < D; ++i) acc += wrow[i] * u_sh[i];
    ubias[tid] = acc;
  }
  __syncthreads();

  // Per-lane loop-invariant bias/W2 over the accumulator VGPR axis.
  float ubr[8], w2r[8];
#pragma unroll
  for (int r = 0; r < 8; ++r) {
    ubr[r] = ubias[cat * H + kb8 + r];
    w2r[r] = W2[cat * H + kb8 + r];
  }
  const float b2c = b2[cat];

  // ---------------- phase 3: scores via WMMA, transposed: D[n, l] ----------
  for (int t = 0; t < NT; ++t) {
    const unsigned short* erow = &eb[(t * 16 + hl) * D + kb8];
    Frag b0, b1f;
    b0.q[0]  = *(const uint4*)(erow);
    b0.q[1]  = *(const uint4*)(erow + 16);
    b1f.q[0] = *(const uint4*)(erow + 32);
    b1f.q[1] = *(const uint4*)(erow + 48);

    v8f acc = {0.f, 0.f, 0.f, 0.f, 0.f, 0.f, 0.f, 0.f};
    acc = __builtin_amdgcn_wmma_f32_16x16x32_bf16(
        false, af0, false, __builtin_bit_cast(v16bf, b0.v),  (short)0, acc, false, false);
    acc = __builtin_amdgcn_wmma_f32_16x16x32_bf16(
        false, af1, false, __builtin_bit_cast(v16bf, b1f.v), (short)0, acc, false, false);

    float sp = 0.f;
#pragma unroll
    for (int r = 0; r < 8; ++r) {
      float h = acc[r] + ubr[r];
      h = h > 0.f ? h : 0.f;
      sp = fmaf(h, w2r[r], sp);
    }
    sp += __shfl_xor(sp, 16, 32);                // combine n=0..7 with n=8..15
    if (lane < 16) s_sh[cat][t * 16 + lane] = sp + b2c;
  }

  // ---------------- phase 4: masked softmax -> bf16 weight matrix ----------
  float sv[KT], zv[KT];
  bool  mem[KT];
  float m = -INFINITY;
#pragma unroll
  for (int k = 0; k < KT; ++k) {
    const int l = lane + 32 * k;
    const bool ok = (catl[l] == cat);
    mem[k] = ok;
    sv[k]  = ok ? s_sh[cat][l] : -INFINITY;
    m = fmaxf(m, sv[k]);
  }
  for (int off = 16; off; off >>= 1) m = fmaxf(m, __shfl_xor(m, off, 32));
  float zs = 0.f;
#pragma unroll
  for (int k = 0; k < KT; ++k) {
    zv[k] = mem[k] ? __expf(sv[k] - m) : 0.f;
    zs += zv[k];
  }
  for (int off = 16; off; off >>= 1) zs += __shfl_xor(zs, off, 32);
  const float inv = 1.f / fmaxf(zs, 1e-30f);
#pragma unroll
  for (int k = 0; k < KT; ++k)
    wb[cat * LP + lane + 32 * k] = f2bf(zv[k] * inv);
  const bool ghas = zs > 0.f;
  __syncthreads();   // all categories' weight rows ready

  // ---------------- phase 5a: gvec = w @ e via WMMA (waves 0..3) -----------
  if (wid < 4) {
    const int ntile = wid;                          // 16 columns of d
    const unsigned short* arow = &wb[hl * LP + kb8];
    const unsigned short* brow = &ebT[(ntile * 16 + hl) * LP + kb8];
    v8f acc = {0.f, 0.f, 0.f, 0.f, 0.f, 0.f, 0.f, 0.f};
#pragma unroll
    for (int ks = 0; ks < KT; ++ks) {
      Frag a, bf;
      a.q[0]  = *(const uint4*)(arow + 32 * ks);        // A[m=cat][k=l]
      a.q[1]  = *(const uint4*)(arow + 32 * ks + 16);
      bf.q[0] = *(const uint4*)(brow + 32 * ks);        // B[k=l][n=d]
      bf.q[1] = *(const uint4*)(brow + 32 * ks + 16);
      acc = __builtin_amdgcn_wmma_f32_16x16x32_bf16(
          false, __builtin_bit_cast(v16bf, a.v), false,
          __builtin_bit_cast(v16bf, bf.v), (short)0, acc, false, false);
    }
    const int d = ntile * 16 + hl;
    if (lane < 16) {                                // rows M=0..7 -> cats 0..7
#pragma unroll
      for (int r = 0; r < 8; ++r) gvec[r][d] = acc[r];
    } else {                                        // row M=8 -> cat 8
      gvec[8][d] = acc[0];
    }
  }
  __syncthreads();

  // ---------------- phase 5b: top-level MLP score per category -------------
  if (lane < 16) {
    const float* wr = Wt1 + lane * (2 * D);
    float acc = bt1[lane];
#pragma unroll 8
    for (int i = 0; i < D; ++i) acc += wr[i] * u_sh[i];
#pragma unroll 8
    for (int i = 0; i < D; ++i) acc += wr[D + i] * gvec[cat][i];
    acc = acc > 0.f ? acc : 0.f;
    float v = acc * Wt2[lane];
    v += __shfl_xor(v, 1, 16);
    v += __shfl_xor(v, 2, 16);
    v += __shfl_xor(v, 4, 16);
    v += __shfl_xor(v, 8, 16);
    if (lane == 0) { s2_sh[cat] = v + bt2[0]; has_sh[cat] = ghas ? 1 : 0; }
  }
  __syncthreads();

  // ---------------- phase 6: softmax over groups + final dot (wave 0) ------
  if (wid == 0) {
    bool ok = false;
    float s2 = -INFINITY;
    if (lane < 16) {
      ok = (lane < C) && (has_sh[lane] != 0);
      s2 = ok ? s2_sh[lane] : -INFINITY;
    }
    float m2 = s2;
    for (int off = 8; off; off >>= 1) m2 = fmaxf(m2, __shfl_xor(m2, off, 16));
    float z2 = ok ? __expf(s2 - m2) : 0.f;
    float zs2 = z2;
    for (int off = 8; off; off >>= 1) zs2 += __shfl_xor(zs2, off, 16);
    if (lane < C) w2_sh[lane] = z2 * (1.f / fmaxf(zs2, 1e-30f));

    float hy0 = 0.f, hy1 = 0.f;
#pragma unroll
    for (int c2 = 0; c2 < C; ++c2) {
      const float wc = w2_sh[c2];
      hy0 += wc * gvec[c2][2 * lane];
      hy1 += wc * gvec[c2][2 * lane + 1];
    }
    const float2 tv = *(const float2*)(item_emb + (long)item_in[b] * D + 2 * lane);
    float p = hy0 * tv.x + hy1 * tv.y;
    for (int off = 16; off; off >>= 1) p += __shfl_xor(p, off, 32);
    if (lane == 0) out[b] = p;
  }
}

extern "C" void kernel_launch(void* const* d_in, const int* in_sizes, int n_in,
                              void* d_out, int out_size, void* d_ws, size_t ws_size,
                              hipStream_t stream) {
  const int*   user_in  = (const int*)d_in[0];
  const int*   rec_in   = (const int*)d_in[1];
  const int*   item_in  = (const int*)d_in[2];
  const int*   item_cat = (const int*)d_in[3];
  const float* user_emb = (const float*)d_in[4];
  const float* item_emb = (const float*)d_in[5];
  const float* W1  = (const float*)d_in[6];
  const float* b1  = (const float*)d_in[7];
  const float* W2  = (const float*)d_in[8];
  const float* b2  = (const float*)d_in[9];
  const float* Wt1 = (const float*)d_in[10];
  const float* bt1 = (const float*)d_in[11];
  const float* Wt2 = (const float*)d_in[12];
  const float* bt2 = (const float*)d_in[13];
  float* out = (float*)d_out;

  const int B = in_sizes[0];   // 2048
  din_fused_kernel<<<B, NTHREADS, 0, stream>>>(
      user_in, rec_in, item_in, item_cat, user_emb, item_emb,
      W1, b1, W2, b2, Wt1, bt1, Wt2, bt2, out);
}
